// ZeroShotModel_10239202034116
// MI455X (gfx1250) — matrix operand order repaired and checked by reading, not compiled, not run on hardware
//
#include <hip/hip_runtime.h>
#include <hip/hip_bf16.h>

typedef __attribute__((ext_vector_type(16))) _Float16 v16h;
typedef __attribute__((ext_vector_type(8)))  _Float16 v8h;
typedef __attribute__((ext_vector_type(4)))  _Float16 v4h;
typedef __attribute__((ext_vector_type(8)))  float    v8f;

#define HDIM 128

// ---- CDNA5 WMMA 16-bit fragment index helpers (CDNA5 ISA §7.12.2) ----
__device__ __forceinline__ int a_kidx(int lane, int e) {
    return ((e & 8) ? 16 : 0) + ((lane & 16) ? 8 : 0) + (e & 7);
}
__device__ __forceinline__ int b_kidx(int lane, int e) {
    return ((lane & 16) ? 16 : 0) + e;
}
// C/D f32 16x16: lane l, vgpr r -> N=(l&15), M=(l>>4)*8 + r

__device__ __forceinline__ v16h load_frag(const _Float16* __restrict__ p,
                                          int fragIdx, int lane) {
    return *(const v16h*)(p + ((size_t)fragIdx * 32 + lane) * 16);
}

__device__ __forceinline__ v8f wmma_f16(v16h a, v16h b, v8f c) {
    return __builtin_amdgcn_wmma_f32_16x16x32_f16(false, a, false, b,
                                                  (short)0, c, false, false);
}

// ---- CDNA5 async Global<->LDS copies (CDNA5 ISA) ----
// GVS mode: mem_addr = SADDR(u64) + VADDR(i32) + IOFFSET; LDS addr from VGPR.
__device__ __forceinline__ unsigned lds_off(const void* p) {
    // addrspacecast(LDS->flat) keeps the LDS byte offset in the low 32 bits
    return (unsigned)(unsigned long long)(uintptr_t)p;
}
__device__ __forceinline__ void async_load_b128(unsigned lds, const void* base,
                                                unsigned voff) {
    asm volatile("global_load_async_to_lds_b128 %0, %1, %2"
                 :: "v"(lds), "v"(voff),
                    "s"((unsigned long long)(uintptr_t)base)
                 : "memory");
}
__device__ __forceinline__ void async_store_b128(unsigned lds, void* base,
                                                 unsigned voff) {
    asm volatile("global_store_async_from_lds_b128 %0, %1, %2"
                 :: "v"(voff), "v"(lds),
                    "s"((unsigned long long)(uintptr_t)base)
                 : "memory");
}
__device__ __forceinline__ void wait_async0() {
    asm volatile("s_wait_asynccnt 0x0" ::: "memory");
}

// =====================================================================
// Weight repack: row-major f32 [K,128] -> fragment-ordered f16.
// =====================================================================
__global__ void pack_weights_kernel(const float* __restrict__ W,
                                    _Float16* __restrict__ out, int K)
{
    int i = blockIdx.x * blockDim.x + threadIdx.x;
    if (i >= K * 128) return;
    int e    = i & 15;
    int lane = (i >> 4) & 31;
    int ct   = (i >> 9) & 7;
    int kb   = i >> 12;
    int k    = kb * 32 + ((e & 8) ? 16 : 0) + ((lane & 16) ? 8 : 0) + (e & 7);
    int col  = ct * 16 + (lane & 15);
    out[i] = (_Float16)W[(size_t)k * 128 + col];
}

// =====================================================================
// Encoder: h(f16) = relu(x[N,32](f32) @ W + b).  K=32 -> one WMMA/tile.
// 64 rows/block, 8 waves. Output staged f16 in LDS -> async store to HBM.
// =====================================================================
__global__ __launch_bounds__(256) void encode_kernel(
    const float* __restrict__ x, const _Float16* __restrict__ Wp,
    const float* __restrict__ bias, _Float16* __restrict__ h, int N)
{
    __shared__ _Float16 xs[64 * 32];     // 4 KB
    __shared__ _Float16 outH[64 * 128];  // 16 KB

    const int tid = threadIdx.x;
    const int rowBase = blockIdx.x * 64;
    const bool full = (rowBase + 64 <= N);

    // stage x (f32 -> f16), float4 loads, 8B packed LDS stores
    for (int i = tid; i < 512; i += 256) {
        int r = i >> 3;
        int row = rowBase + r;
        float4 v = make_float4(0.f, 0.f, 0.f, 0.f);
        if (full || row < N)
            v = ((const float4*)(x + (size_t)row * 32))[i & 7];
        v4h p; p[0] = (_Float16)v.x; p[1] = (_Float16)v.y;
               p[2] = (_Float16)v.z; p[3] = (_Float16)v.w;
        *(v4h*)&xs[r * 32 + (i & 7) * 4] = p;
    }
    __syncthreads();

    const int wave = tid >> 5, lane = tid & 31;
    const int rl = lane & 15, hi = lane >> 4;
    const int rowSub = wave & 3, ctBase = (wave >> 2) * 4;

    v16h a;
#pragma unroll
    for (int e = 0; e < 16; ++e)
        a[e] = xs[(rowSub * 16 + rl) * 32 + a_kidx(lane, e)];

#pragma unroll
    for (int j = 0; j < 4; ++j) {
        const int ct = ctBase + j, col = ct * 16 + rl;
        v16h b = load_frag(Wp, ct, lane);
        v8f acc = {};
        acc = wmma_f16(a, b, acc);
        const float bv = bias[col];
#pragma unroll
        for (int r = 0; r < 8; ++r) {
            float v = acc[r] + bv;
            outH[(rowSub * 16 + hi * 8 + r) * 128 + col] =
                (_Float16)(v > 0.f ? v : 0.f);
        }
    }
    __syncthreads();

    // async store: 64 rows x 256B = 1024 x b128 chunks
    const unsigned lbase = lds_off(&outH[0]);
    for (int i = tid; i < 1024; i += 256) {
        int row = rowBase + (i >> 4);
        if (full || row < N)
            async_store_b128(lbase + i * 16, (void*)h,
                             (unsigned)rowBase * 256 + (unsigned)i * 16);
    }
    // S_ENDPGM performs an implicit wait-idle covering ASYNCcnt.
}

// =====================================================================
// Zero-fill (float4 grid-stride) for the f32 message buffer.
// =====================================================================
__global__ void zero_kernel(float4* __restrict__ p, long n4)
{
    long i  = (long)blockIdx.x * blockDim.x + threadIdx.x;
    long st = (long)gridDim.x * blockDim.x;
    for (; i < n4; i += st) p[i] = make_float4(0.f, 0.f, 0.f, 0.f);
}

// =====================================================================
// Segment-sum: msg(f32)[dst[e],:] += h_src(f16)[src[e],:].
// One wave per edge: b64 f16 load per lane + 4 hw f32 atomics.
// =====================================================================
__global__ __launch_bounds__(256) void scatter_add_kernel(
    const _Float16* __restrict__ hsrc, const int* __restrict__ src,
    const int* __restrict__ dst, float* __restrict__ msg, int E)
{
    const int lane   = threadIdx.x & 31;
    long warp   = ((long)blockIdx.x * blockDim.x + threadIdx.x) >> 5;
    long nwarps = ((long)gridDim.x * blockDim.x) >> 5;
    for (long e = warp; e < E; e += nwarps) {
        const int s = src[e], d = dst[e];
        v4h v = *(const v4h*)(hsrc + (size_t)s * HDIM + lane * 4);
        float* mp = msg + (size_t)d * HDIM + lane * 4;
        atomicAdd(&mp[0], (float)v[0]);
        atomicAdd(&mp[1], (float)v[1]);
        atomicAdd(&mp[2], (float)v[2]);
        atomicAdd(&mp[3], (float)v[3]);
    }
}

// =====================================================================
// Tree-agg MLP (fused, in-place on h(f16)):
//   z = relu([h | msg] @ W1 + b1);  h' = relu(z @ W2 + b2)
// h-half of A staged via async Global->LDS copy; msg-half converted f32->f16.
// Output staged f16 (aliases dead A-stage) -> async LDS->Global store.
// =====================================================================
__global__ __launch_bounds__(256) void tree_kernel(
    const float* __restrict__ msg,
    const _Float16* __restrict__ W1p, const float* __restrict__ b1,
    const _Float16* __restrict__ W2p, const float* __restrict__ b2,
    _Float16* __restrict__ hbuf, int N)
{
    __shared__ _Float16 As[64 * 256];  // 32 KB; first 16 KB reused as out-stage
    __shared__ _Float16 Zs[64 * 128];  // 16 KB
    _Float16* outH = As;               // alias: As dead after GEMM1

    const int tid = threadIdx.x;
    const int rowBase = blockIdx.x * 64;
    const bool full = (rowBase + 64 <= N);
    const unsigned lAs = lds_off(&As[0]);

    // --- stage h-half: async copy, row r -> LDS bytes [r*512, r*512+256) ---
    for (int i = tid; i < 1024; i += 256) {
        int r = i >> 4, j = i & 15;
        int row = rowBase + r;
        if (full || row < N) {
            async_load_b128(lAs + r * 512 + j * 16, (const void*)hbuf,
                            (unsigned)row * 256 + (unsigned)j * 16);
        } else {
            *(v8h*)&As[r * 256 + j * 8] = (v8h)(_Float16)0.0f;
        }
    }
    // --- stage msg-half: f32 -> f16 convert ---
    for (int i = tid; i < 2048; i += 256) {   // 64*128/4 float4 chunks
        int r = i >> 5;
        int row = rowBase + r;
        float4 v = make_float4(0.f, 0.f, 0.f, 0.f);
        if (full || row < N)
            v = ((const float4*)(msg + (size_t)row * HDIM))[i & 31];
        v4h p; p[0] = (_Float16)v.x; p[1] = (_Float16)v.y;
               p[2] = (_Float16)v.z; p[3] = (_Float16)v.w;
        *(v4h*)&As[r * 256 + 128 + (i & 31) * 4] = p;
    }
    wait_async0();
    __syncthreads();

    const int wave = tid >> 5, lane = tid & 31;
    const int rl = lane & 15, hi = lane >> 4;
    const int rowSub = wave & 3, ctBase = (wave >> 2) * 4;

    // ---------------- GEMM1: K = 256 ----------------
    for (int j = 0; j < 4; ++j) {
        const int ct = ctBase + j, col = ct * 16 + rl;
        v8f acc = {};
#pragma unroll
        for (int kb = 0; kb < 8; ++kb) {
            v16h a;
#pragma unroll
            for (int e = 0; e < 16; ++e)
                a[e] = As[(rowSub * 16 + rl) * 256 + kb * 32 + a_kidx(lane, e)];
            v16h b = load_frag(W1p, kb * 8 + ct, lane);
            acc = wmma_f16(a, b, acc);
        }
        const float bv = b1[col];
#pragma unroll
        for (int r = 0; r < 8; ++r) {
            float v = acc[r] + bv;
            Zs[(rowSub * 16 + hi * 8 + r) * 128 + col] = (_Float16)(v > 0.f ? v : 0.f);
        }
    }
    __syncthreads();

    // ---------------- GEMM2: K = 128 -> outH (f16) ----------------
    for (int j = 0; j < 4; ++j) {
        const int ct = ctBase + j, col = ct * 16 + rl;
        v8f acc = {};
#pragma unroll
        for (int kb = 0; kb < 4; ++kb) {
            v16h a;
#pragma unroll
            for (int e = 0; e < 16; ++e)
                a[e] = Zs[(rowSub * 16 + rl) * 128 + kb * 32 + a_kidx(lane, e)];
            v16h b = load_frag(W2p, kb * 8 + ct, lane);
            acc = wmma_f16(a, b, acc);
        }
        const float bv = b2[col];
#pragma unroll
        for (int r = 0; r < 8; ++r) {
            float v = acc[r] + bv;
            outH[(rowSub * 16 + hi * 8 + r) * 128 + col] =
                (_Float16)(v > 0.f ? v : 0.f);
        }
    }
    __syncthreads();

    // --- async write-back: 16 KB, 1024 x b128 chunks ---
    const unsigned lOut = lds_off(&outH[0]);
    for (int i = tid; i < 1024; i += 256) {
        int row = rowBase + (i >> 4);
        if (full || row < N)
            async_store_b128(lOut + i * 16, (void*)hbuf,
                             (unsigned)rowBase * 256 + (unsigned)i * 16);
    }
}

// =====================================================================
// Final head: out(f32) = relu(h_enc(f16) @ Wf1 + bf1) @ Wf2 + bf2
// A tile is a raw f16 copy -> async Global->LDS.
// =====================================================================
__global__ __launch_bounds__(256) void final_kernel(
    const _Float16* __restrict__ henc, const _Float16* __restrict__ Wf1p,
    const float* __restrict__ bf1,  const float* __restrict__ Wf2,
    const float* __restrict__ bf2,  float* __restrict__ out, int N)
{
    __shared__ _Float16 As[64 * 128];  // 16 KB
    __shared__ float    Ts[64 * 128];  // 32 KB

    const int tid = threadIdx.x;
    const int rowBase = blockIdx.x * 64;
    const bool full = (rowBase + 64 <= N);
    const unsigned lAs = lds_off(&As[0]);

    for (int i = tid; i < 1024; i += 256) {
        int row = rowBase + (i >> 4);
        if (full || row < N) {
            async_load_b128(lAs + i * 16, (const void*)henc,
                            (unsigned)rowBase * 256 + (unsigned)i * 16);
        } else {
            *(v8h*)&As[i * 8] = (v8h)(_Float16)0.0f;
        }
    }
    wait_async0();
    __syncthreads();

    const int wave = tid >> 5, lane = tid & 31;
    const int rl = lane & 15, hi = lane >> 4;
    const int rowSub = wave & 3, ctBase = (wave >> 2) * 4;

    for (int j = 0; j < 4; ++j) {
        const int ct = ctBase + j, col = ct * 16 + rl;
        v8f acc = {};
#pragma unroll
        for (int kb = 0; kb < 4; ++kb) {
            v16h a;
#pragma unroll
            for (int e = 0; e < 16; ++e)
                a[e] = As[(rowSub * 16 + rl) * 128 + kb * 32 + a_kidx(lane, e)];
            v16h b = load_frag(Wf1p, kb * 8 + ct, lane);
            acc = wmma_f16(a, b, acc);
        }
        const float bv = bf1[col];
#pragma unroll
        for (int r = 0; r < 8; ++r) {
            float v = acc[r] + bv;
            Ts[(rowSub * 16 + hi * 8 + r) * 128 + col] = v > 0.f ? v : 0.f;
        }
    }
    __syncthreads();

    if (tid < 64) {
        int row = rowBase + tid;
        if (row < N) {
            float s = bf2[0];
#pragma unroll 8
            for (int c = 0; c < 128; ++c) s += Ts[tid * 128 + c] * Wf2[c];
            out[row] = s;
        }
    }
}

// =====================================================================
// Host-side orchestration (all on `stream`, graph-capture safe).
// =====================================================================
extern "C" void kernel_launch(void* const* d_in, const int* in_sizes, int n_in,
                              void* d_out, int out_size, void* d_ws, size_t ws_size,
                              hipStream_t stream)
{
    (void)n_in; (void)out_size; (void)ws_size;

    const float* xs[6];
    for (int i = 0; i < 6; ++i) xs[i] = (const float*)d_in[i];
    const float* W_enc = (const float*)d_in[6];
    const float* b_enc = (const float*)d_in[7];
    const float* W1t   = (const float*)d_in[8];
    const float* b1t   = (const float*)d_in[9];
    const float* W2t   = (const float*)d_in[10];
    const float* b2t   = (const float*)d_in[11];
    const float* Wf1   = (const float*)d_in[12];
    const float* bf1   = (const float*)d_in[13];
    const float* Wf2   = (const float*)d_in[14];
    const float* bf2   = (const float*)d_in[15];
    const int* srcI[5] = {(const int*)d_in[16], (const int*)d_in[18],
                          (const int*)d_in[20], (const int*)d_in[22],
                          (const int*)d_in[24]};
    const int* dstI[5] = {(const int*)d_in[17], (const int*)d_in[19],
                          (const int*)d_in[21], (const int*)d_in[23],
                          (const int*)d_in[25]};
    int E[5];
    for (int s = 0; s < 5; ++s) E[s] = in_sizes[16 + 2 * s];

    int N[6];
    for (int i = 0; i < 6; ++i) N[i] = in_sizes[i] / 32;

    // ---- workspace layout: f16 hidden states, f32 msg, packed f16 weights ----
    char* ws = (char*)d_ws;
    _Float16* h[6];
    size_t off = 0;
    for (int i = 0; i < 6; ++i) {
        h[i] = (_Float16*)(ws + off);
        off += (size_t)N[i] * HDIM * sizeof(_Float16);
    }
    off = (off + 255) & ~(size_t)255;
    float* msg = (float*)(ws + off);
    off += (size_t)N[1] * HDIM * sizeof(float);
    _Float16* Wenc_p[6];
    for (int i = 0; i < 6; ++i) { Wenc_p[i] = (_Float16*)(ws + off); off += 32 * 128 * 2; }
    _Float16* W1p[5];
    for (int s = 0; s < 5; ++s) { W1p[s] = (_Float16*)(ws + off); off += 256 * 128 * 2; }
    _Float16* W2p[5];
    for (int s = 0; s < 5; ++s) { W2p[s] = (_Float16*)(ws + off); off += 128 * 128 * 2; }
    _Float16* Wf1p = (_Float16*)(ws + off); off += 128 * 128 * 2;

    // ---- 0) repack weights to fragment-ordered f16 ----
    for (int i = 0; i < 6; ++i)
        pack_weights_kernel<<<(32 * 128 + 255) / 256, 256, 0, stream>>>(
            W_enc + (size_t)i * 32 * 128, Wenc_p[i], 32);
    for (int s = 0; s < 5; ++s) {
        pack_weights_kernel<<<(256 * 128 + 255) / 256, 256, 0, stream>>>(
            W1t + (size_t)s * 256 * 128, W1p[s], 256);
        pack_weights_kernel<<<(128 * 128 + 255) / 256, 256, 0, stream>>>(
            W2t + (size_t)s * 128 * 128, W2p[s], 128);
    }
    pack_weights_kernel<<<(128 * 128 + 255) / 256, 256, 0, stream>>>(Wf1, Wf1p, 128);

    // ---- 1) node-type encoders ----
    for (int i = 0; i < 6; ++i)
        encode_kernel<<<(N[i] + 63) / 64, 256, 0, stream>>>(
            xs[i], Wenc_p[i], b_enc + (size_t)i * HDIM, h[i], N[i]);

    // ---- 2) five bottom-up message-passing stages ----
    for (int s = 0; s < 5; ++s) {
        const int Ndst = N[s + 1];
        long n4 = (long)Ndst * HDIM / 4;
        long zb = (n4 + 255) / 256; if (zb > 16384) zb = 16384;
        zero_kernel<<<(int)zb, 256, 0, stream>>>((float4*)msg, n4);

        long sb = ((long)E[s] + 7) / 8; if (sb > 32768) sb = 32768;
        scatter_add_kernel<<<(int)sb, 256, 0, stream>>>(h[s], srcI[s], dstI[s], msg, E[s]);

        tree_kernel<<<(Ndst + 63) / 64, 256, 0, stream>>>(
            msg, W1p[s], b1t + (size_t)s * HDIM,
            W2p[s], b2t + (size_t)s * HDIM, h[s + 1], Ndst);
    }

    // ---- 3) final MLP head ----
    final_kernel<<<(N[5] + 63) / 64, 256, 0, stream>>>(
        h[5], Wf1p, bf1, Wf2, bf2, (float*)d_out, N[5]);
}
